// ResUnet_Position_Attention_27436251087205
// MI455X (gfx1250) — matrix operand order repaired and checked by reading, not compiled, not run on hardware
//
#include <hip/hip_runtime.h>
#include <stdint.h>

#define N_TOK 8192
#define C_DIM 256
#define BN_EPS 1e-5f

typedef __attribute__((ext_vector_type(16))) __bf16 v16bf;
typedef __attribute__((ext_vector_type(8)))  float  v8f;
typedef __attribute__((ext_vector_type(4)))  int    v4i;

union FragB16 { uint32_t u[8]; uint16_t s[16]; v16bf v; };

#if defined(__has_builtin)
#if __has_builtin(__builtin_amdgcn_global_load_async_to_lds_b128)
#define HAVE_ASYNC_LDS 1
#endif
#endif

__device__ __forceinline__ uint16_t f2bf(float f) {
  uint32_t u; __builtin_memcpy(&u, &f, 4);
  u += 0x7FFFu + ((u >> 16) & 1u);          // round-to-nearest-even
  return (uint16_t)(u >> 16);
}

__device__ __forceinline__ v8f wmma_bf16(v16bf a, v16bf b, v8f c) {
  // (neg_a, A, neg_b, B, c_mod, C, reuse_a, reuse_b)
  return __builtin_amdgcn_wmma_f32_16x16x32_bf16(false, a, false, b, (short)0, c,
                                                 false, false);
}

// A-fragment (16x32 bf16, row-major source): lane l (m=l&15, g=l>>4)
// halfwords 0..15 = K: k0 + 16g + 0..15
__device__ __forceinline__ v16bf load_a_frag(const uint16_t* rowp /* row base + k0 */,
                                             int g) {
  FragB16 f;
  const uint32_t* p0 = (const uint32_t*)(rowp + 16 * g);
  const uint32_t* p1 = (const uint32_t*)(rowp + 16 * g + 8);
  f.u[0] = p0[0]; f.u[1] = p0[1]; f.u[2] = p0[2]; f.u[3] = p0[3];
  f.u[4] = p1[0]; f.u[5] = p1[1]; f.u[6] = p1[2]; f.u[7] = p1[3];
  return f.v;
}

// B-fragment (32x16 bf16) taken as M^T of a row-major M (column n of B = row n of
// M). Lane l (n=l&15, g=l>>4) holds K = k0 + 16g + 0..15 -> one contiguous 32B.
__device__ __forceinline__ v16bf load_bt_frag(const uint16_t* p /* M row + k0 + 16g */) {
  FragB16 f;
  const uint32_t* q = (const uint32_t*)p;
#pragma unroll
  for (int i = 0; i < 8; ++i) f.u[i] = q[i];
  return f.v;
}

// ---------------------------------------------------------------- convert f32->bf16
__global__ void cvt_bf16_kernel(const float* __restrict__ in,
                                uint16_t* __restrict__ out, int n) {
  int i = blockIdx.x * blockDim.x + threadIdx.x;
  if (i < n) out[i] = f2bf(in[i]);
}

// ---------------------------------------------------------------- Y = X * W^T + b
// X: [rows, C] bf16, W: [C, C] bf16, bias: [C] f32. 4 waves, 64x64 block tile.
template <bool WF, bool WB>
__global__ __launch_bounds__(128)
void gemm_xwt_kernel(const uint16_t* __restrict__ A, const uint16_t* __restrict__ W,
                     const float* __restrict__ bias, float* __restrict__ outF,
                     uint16_t* __restrict__ outB) {
  const int lane = threadIdx.x & 31;
  const int wave = threadIdx.x >> 5;
  const int g = lane >> 4;
  const int ln = lane & 15;
  const int r0 = blockIdx.y * 64 + wave * 16;
  const int c0 = blockIdx.x * 64;

  const v8f vzero = {0.f, 0.f, 0.f, 0.f, 0.f, 0.f, 0.f, 0.f};
  v8f acc[4] = {vzero, vzero, vzero, vzero};

  const uint16_t* arow = A + (size_t)(r0 + ln) * C_DIM;
#pragma unroll 2
  for (int kt = 0; kt < C_DIM / 32; ++kt) {
    const int k0 = kt * 32;
    __builtin_prefetch(arow + k0 + 64, 0, 0);   // global_prefetch_b8
    v16bf a = load_a_frag(arow + k0, g);
#pragma unroll
    for (int t = 0; t < 4; ++t) {
      const uint16_t* wrow = W + (size_t)(c0 + t * 16 + ln) * C_DIM + k0 + 16 * g;
      acc[t] = wmma_bf16(a, load_bt_frag(wrow), acc[t]);
    }
  }
#pragma unroll
  for (int t = 0; t < 4; ++t) {
    const int cc = c0 + t * 16 + ln;
    const float bv = bias[cc];
#pragma unroll
    for (int i = 0; i < 8; ++i) {
      const int rr = r0 + i + 8 * g;                 // C/D layout: M = i (+8 hi half)
      const float val = acc[t][i] + bv;
      const size_t idx = (size_t)rr * C_DIM + cc;
      if (WF) outF[idx] = val;
      if (WB) outB[idx] = f2bf(val);
    }
  }
}

// ---------------------------------------------------------------- bf16 transpose
// in: [N,C] -> out: [C,N], 64x64 tiles through LDS. Load phase uses CDNA5
// async global->LDS (ASYNCcnt-tracked) when the builtin is available.
__global__ __launch_bounds__(256)
void transpose_bf16_kernel(const uint16_t* __restrict__ in,
                           uint16_t* __restrict__ out) {
  __shared__ uint16_t tile[64][72];   // 144B row stride: 16B-aligned rows
  const int r0 = blockIdx.x * 64;     // source rows (keys)
  const int c0 = blockIdx.y * 64;     // source cols (channels)
  const int t = threadIdx.x;

#if defined(HAVE_ASYNC_LDS)
#pragma unroll
  for (int i = 0; i < 2; ++i) {       // 512 chunks of 8 bf16, 2 per thread
    const int idx = t + 256 * i;
    const int r = idx >> 3;
    const int ch = idx & 7;
    const uint16_t* g = in + (size_t)(r0 + r) * C_DIM + c0 + ch * 8;
    uint16_t* l = &tile[r][ch * 8];
    __builtin_amdgcn_global_load_async_to_lds_b128(
        (__attribute__((address_space(1))) v4i*)const_cast<uint16_t*>(g),
        (__attribute__((address_space(3))) v4i*)l, 0, 0);
  }
#if __has_builtin(__builtin_amdgcn_s_wait_asynccnt)
  __builtin_amdgcn_s_wait_asynccnt(0);
#else
  asm volatile("s_wait_asynccnt 0x0" ::: "memory");
#endif
  __syncthreads();
#else
#pragma unroll
  for (int i = 0; i < 8; ++i) {       // 2048 uint chunks, 8 per thread
    const int idx = t + 256 * i;
    const int r = idx >> 5;
    const int cp = idx & 31;
    *(uint32_t*)&tile[r][cp * 2] =
        *(const uint32_t*)(in + (size_t)(r0 + r) * C_DIM + c0 + cp * 2);
  }
  __syncthreads();
#endif

#pragma unroll
  for (int i = 0; i < 8; ++i) {       // write out^T, 2 keys per uint store
    const int idx = t + 256 * i;
    const int c = idx >> 5;
    const int kp = idx & 31;
    const uint32_t lo = tile[kp * 2][c];
    const uint32_t hi = tile[kp * 2 + 1][c];
    *(uint32_t*)(out + (size_t)(c0 + c) * N_TOK + r0 + kp * 2) = lo | (hi << 16);
  }
}

// ---------------------------------------------------------------- flash attention
// xr = softmax(q k^T / 16) v - v.  One block = 16 query rows, 4 waves.
// Wave w: partial S over channels [64w,64w+64); owns output cols [64w,64w+64).
// Vt is V transposed ([C,N]) so P@V B-fragments are contiguous 32B loads.
__global__ __launch_bounds__(128)
void flash_attn_kernel(const uint16_t* __restrict__ Q, const uint16_t* __restrict__ K,
                       const uint16_t* __restrict__ Vt, const float* __restrict__ Vf,
                       uint16_t* __restrict__ xrB) {
  __shared__ float sS[4][16 * 32];
  const int lane = threadIdx.x & 31;
  const int wave = threadIdx.x >> 5;
  const int g = lane >> 4;
  const int ln = lane & 15;
  const int q0 = blockIdx.x * 16;
  const int ch0 = wave * 64;

  const uint16_t* qrow = Q + (size_t)(q0 + ln) * C_DIM + ch0;
  const v16bf aq0 = load_a_frag(qrow, g);
  const v16bf aq1 = load_a_frag(qrow + 32, g);

  const v8f vzero = {0.f, 0.f, 0.f, 0.f, 0.f, 0.f, 0.f, 0.f};
  v8f o[4] = {vzero, vzero, vzero, vzero};
  float Mst[8], Lst[8];
#pragma unroll
  for (int i = 0; i < 8; ++i) { Mst[i] = -__builtin_inff(); Lst[i] = 0.f; }
  const float scale = 0.0625f;  // 1/sqrt(256)

  for (int kb = 0; kb < N_TOK / 32; ++kb) {
    const int key0 = kb * 32;
    // ---- partial S (16x32) over this wave's 64 channels: 4 WMMA
    v8f s0 = vzero, s1 = vzero;
    {
      const uint16_t* k_n0 = K + (size_t)(key0 + ln) * C_DIM + ch0 + 16 * g;
      const uint16_t* k_n1 = K + (size_t)(key0 + 16 + ln) * C_DIM + ch0 + 16 * g;
      s0 = wmma_bf16(aq0, load_bt_frag(k_n0), s0);
      s1 = wmma_bf16(aq0, load_bt_frag(k_n1), s1);
      s0 = wmma_bf16(aq1, load_bt_frag(k_n0 + 32), s0);
      s1 = wmma_bf16(aq1, load_bt_frag(k_n1 + 32), s1);
    }
    // ---- cross-wave reduce via LDS
#pragma unroll
    for (int i = 0; i < 8; ++i) {
      const int row = i + 8 * g;
      sS[wave][row * 32 + ln] = s0[i];
      sS[wave][row * 32 + 16 + ln] = s1[i];
    }
    __syncthreads();
#pragma unroll
    for (int i = 0; i < 8; ++i) {
      const int row = i + 8 * g;
      float t0 = sS[0][row * 32 + ln] + sS[1][row * 32 + ln] +
                 sS[2][row * 32 + ln] + sS[3][row * 32 + ln];
      float t1 = sS[0][row * 32 + 16 + ln] + sS[1][row * 32 + 16 + ln] +
                 sS[2][row * 32 + 16 + ln] + sS[3][row * 32 + 16 + ln];
      s0[i] = t0 * scale;
      s1[i] = t1 * scale;
    }
    __syncthreads();
    // ---- online softmax (row r = i + 8g lives in a fixed 16-lane half)
    float alpha[8];
#pragma unroll
    for (int i = 0; i < 8; ++i) {
      float m = fmaxf(s0[i], s1[i]);
#pragma unroll
      for (int d = 1; d < 16; d <<= 1) m = fmaxf(m, __shfl_xor(m, d, 32));
      const float newM = fmaxf(Mst[i], m);
      alpha[i] = __expf(Mst[i] - newM);
      s0[i] = __expf(s0[i] - newM);
      s1[i] = __expf(s1[i] - newM);
      float rs = s0[i] + s1[i];
#pragma unroll
      for (int d = 1; d < 16; d <<= 1) rs += __shfl_xor(rs, d, 32);
      Lst[i] = Lst[i] * alpha[i] + rs;
      Mst[i] = newM;
    }
#pragma unroll
    for (int t = 0; t < 4; ++t)
#pragma unroll
      for (int i = 0; i < 8; ++i) o[t][i] *= alpha[i];
    // ---- re-layout P: C/D layout -> A fragment, via this wave's LDS slot
#pragma unroll
    for (int i = 0; i < 8; ++i) {
      const int row = i + 8 * g;
      sS[wave][row * 32 + ln] = s0[i];
      sS[wave][row * 32 + 16 + ln] = s1[i];
    }
    FragB16 ap;
#pragma unroll
    for (int j = 0; j < 16; ++j) {
      const int kk = 16 * g + j;
      ap.s[j] = f2bf(sS[wave][ln * 32 + kk]);
    }
    // ---- O += P @ V : B-fragments contiguous from Vt ([C,N]) : 4 WMMA
#pragma unroll
    for (int t = 0; t < 4; ++t) {
      const int c = ch0 + t * 16 + ln;
      const uint16_t* vrow = Vt + (size_t)c * N_TOK + key0 + 16 * g;
      o[t] = wmma_bf16(ap.v, load_bt_frag(vrow), o[t]);
    }
  }
  // ---- finalize: O / L - v, store bf16 for the final linear
  float rl[8];
#pragma unroll
  for (int i = 0; i < 8; ++i) rl[i] = 1.0f / Lst[i];
#pragma unroll
  for (int t = 0; t < 4; ++t) {
    const int c = ch0 + t * 16 + ln;
#pragma unroll
    for (int i = 0; i < 8; ++i) {
      const size_t idx = (size_t)(q0 + i + 8 * g) * C_DIM + c;
      xrB[idx] = f2bf(o[t][i] * rl[i] - Vf[idx]);
    }
  }
}

// ---------------------------------------------------------------- BN helpers
__global__ void zero_f32_kernel(float* p, int n) {
  int i = blockIdx.x * blockDim.x + threadIdx.x;
  if (i < n) p[i] = 0.f;
}

__global__ __launch_bounds__(256)
void bn_reduce_kernel(const float* __restrict__ h, float* __restrict__ mu,
                      float* __restrict__ ms) {
  const int c = threadIdx.x;
  const int r0 = blockIdx.x * 256;
  float s = 0.f, s2 = 0.f;
  for (int r = 0; r < 256; ++r) {
    const float v = h[(size_t)(r0 + r) * C_DIM + c];
    s += v; s2 += v * v;
  }
  atomicAdd(&mu[c], s);
  atomicAdd(&ms[c], s2);
}

__global__ void bn_finalize_kernel(const float* __restrict__ h,
                                   const float* __restrict__ x,
                                   const float* __restrict__ mu,
                                   const float* __restrict__ ms,
                                   const float* __restrict__ gamma,
                                   const float* __restrict__ beta,
                                   float* __restrict__ out) {
  const int c = threadIdx.x;
  const size_t idx = (size_t)blockIdx.x * C_DIM + c;
  const float mean = mu[c] * (1.0f / N_TOK);
  const float var = ms[c] * (1.0f / N_TOK) - mean * mean;
  const float inv = rsqrtf(var + BN_EPS);
  float val = (h[idx] - mean) * inv * gamma[c] + beta[c];
  val = fmaxf(val, 0.f);
  out[idx] = val + x[idx];
}

// ----------------------------------------------------------------
extern "C" void kernel_launch(void* const* d_in, const int* in_sizes, int n_in,
                              void* d_out, int out_size, void* d_ws, size_t ws_size,
                              hipStream_t stream) {
  (void)in_sizes; (void)n_in; (void)out_size; (void)ws_size;
  const float* x   = (const float*)d_in[0];
  const float* Wq  = (const float*)d_in[1];
  const float* bq  = (const float*)d_in[2];
  const float* Wk  = (const float*)d_in[3];
  const float* bk  = (const float*)d_in[4];
  const float* Wv  = (const float*)d_in[5];
  const float* bvv = (const float*)d_in[6];
  const float* Wl  = (const float*)d_in[7];
  const float* bl  = (const float*)d_in[8];
  const float* gamma = (const float*)d_in[9];
  const float* beta  = (const float*)d_in[10];
  float* out = (float*)d_out;

  const size_t NC = (size_t)N_TOK * C_DIM;
  const size_t CC = (size_t)C_DIM * C_DIM;

  char* ws = (char*)d_ws;
  size_t off = 0;
  auto alloc = [&](size_t bytes) -> void* {
    off = (off + 255) & ~(size_t)255;
    void* p = ws + off;
    off += bytes;
    return p;
  };
  uint16_t* xb   = (uint16_t*)alloc(NC * 2);
  uint16_t* wqb  = (uint16_t*)alloc(CC * 2);
  uint16_t* wkb  = (uint16_t*)alloc(CC * 2);
  uint16_t* wvb  = (uint16_t*)alloc(CC * 2);
  uint16_t* wlb  = (uint16_t*)alloc(CC * 2);
  uint16_t* qb   = (uint16_t*)alloc(NC * 2);
  uint16_t* kb   = (uint16_t*)alloc(NC * 2);
  uint16_t* vb   = (uint16_t*)alloc(NC * 2);
  uint16_t* vtb  = (uint16_t*)alloc(NC * 2);
  float*    vf   = (float*)alloc(NC * 4);
  uint16_t* xrb  = (uint16_t*)alloc(NC * 2);
  float*    hbuf = (float*)alloc(NC * 4);
  float*    stats = (float*)alloc(512 * 4);
  float* mu = stats;
  float* ms = stats + 256;

  // 1) f32 -> bf16 conversions
  cvt_bf16_kernel<<<(int)(NC / 256), 256, 0, stream>>>(x,  xb,  (int)NC);
  cvt_bf16_kernel<<<(int)(CC / 256), 256, 0, stream>>>(Wq, wqb, (int)CC);
  cvt_bf16_kernel<<<(int)(CC / 256), 256, 0, stream>>>(Wk, wkb, (int)CC);
  cvt_bf16_kernel<<<(int)(CC / 256), 256, 0, stream>>>(Wv, wvb, (int)CC);
  cvt_bf16_kernel<<<(int)(CC / 256), 256, 0, stream>>>(Wl, wlb, (int)CC);

  // 2) q/k/v projections (WMMA GEMMs)
  dim3 ggrid(C_DIM / 64, N_TOK / 64);
  gemm_xwt_kernel<false, true><<<ggrid, 128, 0, stream>>>(xb, wqb, bq,  nullptr, qb);
  gemm_xwt_kernel<false, true><<<ggrid, 128, 0, stream>>>(xb, wkb, bk,  nullptr, kb);
  gemm_xwt_kernel<true,  true><<<ggrid, 128, 0, stream>>>(xb, wvb, bvv, vf,      vb);

  // 2b) V^T so flash P@V B-fragments are contiguous (async global->LDS staging)
  dim3 tgrid(N_TOK / 64, C_DIM / 64);
  transpose_bf16_kernel<<<tgrid, 256, 0, stream>>>(vb, vtb);

  // 3) flash attention (Q/K/V stay L2-resident), xr = attn@v - v
  flash_attn_kernel<<<N_TOK / 16, 128, 0, stream>>>(qb, kb, vtb, vf, xrb);

  // 4) final linear h = xr @ Wl^T + bl
  gemm_xwt_kernel<true, false><<<ggrid, 128, 0, stream>>>(xrb, wlb, bl, hbuf, nullptr);

  // 5) batch-norm stats (zero every call -> deterministic under graph replay)
  zero_f32_kernel<<<1, 512, 0, stream>>>(stats, 512);
  bn_reduce_kernel<<<N_TOK / 256, 256, 0, stream>>>(hbuf, mu, ms);

  // 6) normalize + ReLU + residual
  bn_finalize_kernel<<<N_TOK, C_DIM, 0, stream>>>(hbuf, x, mu, ms, gamma, beta, out);
}
